// GCNConv_Qnt_87660282511729
// MI455X (gfx1250) — matrix operand order repaired and checked by reading, not compiled, not run on hardware
//
#include <hip/hip_runtime.h>
#include <hip/hip_bf16.h>

// ---------------------------------------------------------------------------
// Quantized GCN (QGTC GCNConv_Qnt) for MI455X / gfx1250, wave32 + WMMA f16.
//
// Exact-integer formulation:
//   Xq_int in [-3,3], Wi_int in [-1,1]  => P1int = Xq_int @ Wi_int, |.|<=384
//   Aq in {0,1}; H = Aq @ P1int, |.| <= 8192*384 = 3.1M < 2^24  (exact in f32)
//   hq_int in [-3,3], Wo_int in [-1,1]  => Gint,  |.|<=192
//   Out = Aq @ Gint, |.| <= 8192*192 = 1.6M < 2^24               (exact in f32)
// All WMMA inputs exactly representable in f16; f32 accumulation exact;
// atomicAdd K-split accumulation of integer-valued floats is associative =>
// deterministic. Scales applied as f32 epilogues.
//
// Bandwidth plan: A (256 MB fp32) is read from HBM exactly ONCE, bit-packed
// via wave32 ballot into 8 MB (L2-resident, 192 MB L2). Both WMMA GEMMs
// decode bits -> f16 {0,1} fragments in VALU (co-executes with XDL WMMA).
// Inner loops are explicit two-stage ping-pong pipelines (no buffer rotation
// movs; loads for stage k+32 overlap WMMAs for stage k).
// ---------------------------------------------------------------------------

typedef __attribute__((ext_vector_type(16))) _Float16 v16h;
typedef __attribute__((ext_vector_type(8)))  _Float16 v8h;
typedef __attribute__((ext_vector_type(8)))  float    v8f;

#define NN     8192
#define NW     (NN / 32)   // 256 packed words per row
#define DIN    128
#define DH     64
#define DOUT   16
#define KS1    8      // K-split, layer 1 big GEMM (chunk 1024 = 16 x 64)
#define KS2    8      // K-split, layer 2 big GEMM

// ws layout (byte offsets)
#define SCAL_OFF   0                       // 4 x u32: maxX, maxWi, maxWo, maxHint
#define HACC_OFF   256                     // 8192*64 f32  (2 MB)
#define HACC_N     (NN * DH)
#define OACC_OFF   (HACC_OFF + HACC_N * 4) // 8192*16 f32  (512 KB)
#define OACC_N     (NN * DOUT)
#define P1T_OFF    (OACC_OFF + OACC_N * 4) // 64 x 8192 f16 (1 MB), transposed
#define GT_OFF     (P1T_OFF + DH * NN * 2) // 16 x 8192 f16 (256 KB), transposed
#define APK_OFF    (GT_OFF + DOUT * NN * 2)// 8192 x 256 u32 bit-packed A (8 MB)

// ---------------------------------------------------------------- helpers ---
__device__ __forceinline__ float scale_of(unsigned bits, float qmax) {
  float s = __uint_as_float(bits) / qmax;
  return (s == 0.0f) ? 1.0f : s;           // jnp.where(scale==0, 1, scale)
}
__device__ __forceinline__ float qclip(float v, float s, float qmax) {
  // round-half-even (rintf, default RNE) matching jnp.round, then clip.
  return fminf(qmax, fmaxf(-qmax, rintf(v / s)));
}

// A fragment (16x32 f16, MxK), contiguous K order matching load_b: lane<16
// holds row M=lane, K = base..base+15; lane>=16 holds row M=lane-16,
// K = base+16..base+31.  `w` is the packed 32-bit word covering
// K = base..base+31 of this lane's row; bit k corresponds to column base+k.
__device__ __forceinline__ v16h decode_a(unsigned w, int half) {
  const unsigned lo = w >> (half * 16);       // chunk0 bits at [7:0]
  const unsigned hi = w >> (half * 16 + 8);   // chunk1 bits at [7:0]
  v16h r;
#pragma unroll
  for (int i = 0; i < 8; ++i) {
    r[i]     = ((lo >> i) & 1u) ? (_Float16)1.0f : (_Float16)0.0f;
    r[i + 8] = ((hi >> i) & 1u) ? (_Float16)1.0f : (_Float16)0.0f;
  }
  return r;
}

// B fragment (32x16 f16, KxN), same K order as decode_a: lane<16 holds
// column N=lane with K = base..base+15 contiguous; lane>=16 holds column
// N=lane-16 with K = base+16..base+31. `p` points at 16 contiguous halves.
__device__ __forceinline__ v16h load_b(const _Float16* __restrict__ p) {
  v8h lo = *reinterpret_cast<const v8h*>(p);
  v8h hi = *reinterpret_cast<const v8h*>(p + 8);
  v16h r;
#pragma unroll
  for (int i = 0; i < 8; ++i) { r[i] = lo[i]; r[i + 8] = hi[i]; }
  return r;
}

__device__ __forceinline__ v8f wmma_f16(v16h a, v16h b, v8f c) {
  return __builtin_amdgcn_wmma_f32_16x16x32_f16(
      /*neg_a=*/false, a, /*neg_b=*/false, b,
      /*c_mod=*/(short)0, c, /*reuse_a=*/false, /*reuse_b=*/false);
}

// ------------------------------------------------------------- K0: zeroing --
__global__ void k_zero(float* __restrict__ acc, int n, unsigned* __restrict__ scal) {
  if (blockIdx.x == 0 && threadIdx.x < 4) scal[threadIdx.x] = 0u;
  for (int i = blockIdx.x * blockDim.x + threadIdx.x; i < n;
       i += gridDim.x * blockDim.x)
    acc[i] = 0.0f;
}

// ---------------------- K-pack: bit-pack Aq = (A > 0) via wave32 ballot -----
__global__ void k_pack(const float* __restrict__ A,
                       unsigned* __restrict__ Apk, long n) {
  const long stride = (long)gridDim.x * blockDim.x;   // multiple of 32
  for (long i = blockIdx.x * (long)blockDim.x + threadIdx.x; i < n; i += stride) {
    const unsigned long long m = __ballot(A[i] > 0.0f);  // wave32: low 32 bits
    if ((threadIdx.x & 31) == 0) Apk[i >> 5] = (unsigned)m;
  }
}

// ----------------------------------------------------------- abs-max (>=0) --
__global__ void k_absmax(const float* __restrict__ x, int n,
                         unsigned* __restrict__ out) {
  float m = 0.0f;
  for (int i = blockIdx.x * blockDim.x + threadIdx.x; i < n;
       i += gridDim.x * blockDim.x)
    m = fmaxf(m, fabsf(x[i]));
  __shared__ float red[256];
  red[threadIdx.x] = m;
  __syncthreads();
  for (int s = blockDim.x >> 1; s > 0; s >>= 1) {
    if (threadIdx.x < s)
      red[threadIdx.x] = fmaxf(red[threadIdx.x], red[threadIdx.x + s]);
    __syncthreads();
  }
  if (threadIdx.x == 0) atomicMax(out, __float_as_uint(red[0]));
}

// ------------------------- K1: P1T[j][i] = (Xq_int @ Wi_int)^T, f16 integers -
__global__ void k_p1t(const float* __restrict__ X, const float* __restrict__ Wi,
                      const unsigned* __restrict__ scal,
                      _Float16* __restrict__ P1T) {
  const int i = blockIdx.x;         // node row
  const int t = threadIdx.x;        // 64 threads = DH columns
  const float s_x = scale_of(scal[0], 3.0f);
  const float s_w = scale_of(scal[1], 1.0f);
  __shared__ float xq[DIN];
  xq[t]      = qclip(X[(long)i * DIN + t],      s_x, 3.0f);
  xq[t + 64] = qclip(X[(long)i * DIN + t + 64], s_x, 3.0f);
  __syncthreads();
  float acc = 0.0f;
#pragma unroll 4
  for (int k = 0; k < DIN; ++k)
    acc += xq[k] * qclip(Wi[k * DH + t], s_w, 1.0f);
  P1T[(long)t * NN + i] = (_Float16)acc;   // |acc| <= 384, exact in f16
}

// --------------------------- K2: Hacc += Aq @ P1int  (WMMA, K-split atomic) --
struct Set1 { unsigned aw; v16h b0, b1, b2, b3; };

__device__ __forceinline__ void g1_load(Set1& s, const unsigned* __restrict__ aRow,
                                        const _Float16* __restrict__ bp0,
                                        const _Float16* __restrict__ bp1,
                                        const _Float16* __restrict__ bp2,
                                        const _Float16* __restrict__ bp3,
                                        int k, int hOff) {
  s.aw = aRow[k >> 5];
  s.b0 = load_b(bp0 + k + hOff);
  s.b1 = load_b(bp1 + k + hOff);
  s.b2 = load_b(bp2 + k + hOff);
  s.b3 = load_b(bp3 + k + hOff);
}
__device__ __forceinline__ void g1_comp(const Set1& s, int half,
                                        v8f& a0, v8f& a1, v8f& a2, v8f& a3) {
  const v16h a = decode_a(s.aw, half);
  a0 = wmma_f16(a, s.b0, a0);
  a1 = wmma_f16(a, s.b1, a1);
  a2 = wmma_f16(a, s.b2, a2);
  a3 = wmma_f16(a, s.b3, a3);
}

__global__ __launch_bounds__(256) void k_gemm1(
    const unsigned* __restrict__ Apk, const _Float16* __restrict__ P1T,
    float* __restrict__ Hacc) {
  const int lane = threadIdx.x & 31;
  const int wave = blockIdx.x * (blockDim.x >> 5) + (threadIdx.x >> 5);
  const int rowTile = wave & (NN / 16 - 1);   // 0..511
  const int kChunk  = wave >> 9;              // 0..KS1-1
  const int rowBase = rowTile * 16;
  const int m    = lane & 15;
  const int half = lane >> 4;
  const long row = rowBase + m;
  const int kBeg = kChunk * (NN / KS1);
  const int kEnd = kBeg + (NN / KS1);         // chunk = 1024 = 16 x 64

  const unsigned* aRow = Apk + row * (long)NW;
  const _Float16* bp0 = P1T + (long)(m)      * NN;
  const _Float16* bp1 = P1T + (long)(m + 16) * NN;
  const _Float16* bp2 = P1T + (long)(m + 32) * NN;
  const _Float16* bp3 = P1T + (long)(m + 48) * NN;
  const int hOff = half * 16;

  v8f acc0 = {}, acc1 = {}, acc2 = {}, acc3 = {};
  Set1 s0, s1;
  g1_load(s0, aRow, bp0, bp1, bp2, bp3, kBeg, hOff);
  int kb = kBeg;
  for (; kb < kEnd - 64; kb += 64) {
    g1_load(s1, aRow, bp0, bp1, bp2, bp3, kb + 32, hOff);
    g1_comp(s0, half, acc0, acc1, acc2, acc3);
    g1_load(s0, aRow, bp0, bp1, bp2, bp3, kb + 64, hOff);
    g1_comp(s1, half, acc0, acc1, acc2, acc3);
  }
  g1_load(s1, aRow, bp0, bp1, bp2, bp3, kb + 32, hOff);
  g1_comp(s0, half, acc0, acc1, acc2, acc3);
  g1_comp(s1, half, acc0, acc1, acc2, acc3);

  // C/D layout: VGPR r, lane<16 -> M=r, lane>=16 -> M=r+8; N = lane&15.
#pragma unroll
  for (int r = 0; r < 8; ++r) {
    const long orow = rowBase + r + half * 8;
    atomicAdd(&Hacc[orow * DH +  0 + m], acc0[r]);
    atomicAdd(&Hacc[orow * DH + 16 + m], acc1[r]);
    atomicAdd(&Hacc[orow * DH + 32 + m], acc2[r]);
    atomicAdd(&Hacc[orow * DH + 48 + m], acc3[r]);
  }
}

// ---------------- K4: GT[n][i] = (hq_int @ Wo_int)^T, f16 integers ----------
__global__ void k_gt(const float* __restrict__ Hacc,
                     const float* __restrict__ Wo,
                     const unsigned* __restrict__ scal,
                     _Float16* __restrict__ GT) {
  const int i = blockIdx.x;
  const int t = threadIdx.x;  // 64
  const float s_x  = scale_of(scal[0], 3.0f);
  const float s_wi = scale_of(scal[1], 1.0f);
  const float s1   = s_x * s_wi;
  float s_h = (__uint_as_float(scal[3]) * s1) / 3.0f;  // max|h|/3
  if (s_h == 0.0f) s_h = 1.0f;
  const float s_wo = scale_of(scal[2], 1.0f);
  __shared__ float hq[DH];
  const float h = Hacc[(long)i * DH + t] * s1;
  hq[t] = fminf(3.0f, fmaxf(-3.0f, rintf(h / s_h)));
  __syncthreads();
  if (t < DOUT) {
    float acc = 0.0f;
#pragma unroll 4
    for (int k = 0; k < DH; ++k)
      acc += hq[k] * qclip(Wo[k * DOUT + t], s_wo, 1.0f);
    GT[(long)t * NN + i] = (_Float16)acc;  // |acc| <= 192, exact in f16
  }
}

// --------------------------- K5: Oacc += Aq @ Gint (WMMA, K-split atomic) ---
struct Set2 { unsigned aw; v16h b; };

__device__ __forceinline__ void g2_load(Set2& s, const unsigned* __restrict__ aRow,
                                        const _Float16* __restrict__ bp,
                                        int k, int hOff) {
  s.aw = aRow[k >> 5];
  s.b  = load_b(bp + k + hOff);
}
__device__ __forceinline__ void g2_comp(const Set2& s, int half, v8f& a0) {
  const v16h a = decode_a(s.aw, half);
  a0 = wmma_f16(a, s.b, a0);
}

__global__ __launch_bounds__(256) void k_gemm2(
    const unsigned* __restrict__ Apk, const _Float16* __restrict__ GT,
    float* __restrict__ Oacc) {
  const int lane = threadIdx.x & 31;
  const int wave = blockIdx.x * (blockDim.x >> 5) + (threadIdx.x >> 5);
  const int rowTile = wave & (NN / 16 - 1);
  const int kChunk  = wave >> 9;              // 0..KS2-1
  const int rowBase = rowTile * 16;
  const int m    = lane & 15;
  const int half = lane >> 4;
  const long row = rowBase + m;
  const int kBeg = kChunk * (NN / KS2);
  const int kEnd = kBeg + (NN / KS2);         // chunk = 1024 = 16 x 64

  const unsigned* aRow = Apk + row * (long)NW;
  const _Float16* bp = GT + (long)m * NN;
  const int hOff = half * 16;

  v8f acc = {};
  Set2 s0, s1;
  g2_load(s0, aRow, bp, kBeg, hOff);
  int kb = kBeg;
  for (; kb < kEnd - 64; kb += 64) {
    g2_load(s1, aRow, bp, kb + 32, hOff);
    g2_comp(s0, half, acc);
    g2_load(s0, aRow, bp, kb + 64, hOff);
    g2_comp(s1, half, acc);
  }
  g2_load(s1, aRow, bp, kb + 32, hOff);
  g2_comp(s0, half, acc);
  g2_comp(s1, half, acc);

#pragma unroll
  for (int r = 0; r < 8; ++r) {
    const long orow = rowBase + r + half * 8;
    atomicAdd(&Oacc[orow * DOUT + m], acc[r]);
  }
}

// --------------------------------- K6: scale epilogue into d_out ------------
__global__ void k_finish(const float* __restrict__ Oacc,
                         const unsigned* __restrict__ scal,
                         float* __restrict__ out, int n) {
  const float s_x  = scale_of(scal[0], 3.0f);
  const float s_wi = scale_of(scal[1], 1.0f);
  float s_h = (__uint_as_float(scal[3]) * (s_x * s_wi)) / 3.0f;
  if (s_h == 0.0f) s_h = 1.0f;
  const float s_wo = scale_of(scal[2], 1.0f);
  const float s2 = s_h * s_wo;
  for (int i = blockIdx.x * blockDim.x + threadIdx.x; i < n;
       i += gridDim.x * blockDim.x)
    out[i] = Oacc[i] * s2;
}

// ---------------------------------------------------------------------------
extern "C" void kernel_launch(void* const* d_in, const int* in_sizes, int n_in,
                              void* d_out, int out_size, void* d_ws, size_t ws_size,
                              hipStream_t stream) {
  const float* A  = (const float*)d_in[0];   // 8192 x 8192
  const float* X  = (const float*)d_in[1];   // 8192 x 128
  const float* Wi = (const float*)d_in[2];   // 128 x 64
  const float* Wo = (const float*)d_in[3];   // 64 x 16
  float* out = (float*)d_out;                // 8192 x 16

  char* ws = (char*)d_ws;
  unsigned* scal   = (unsigned*)(ws + SCAL_OFF);
  float* Hacc      = (float*)(ws + HACC_OFF);
  float* Oacc      = (float*)(ws + OACC_OFF);
  _Float16* P1T    = (_Float16*)(ws + P1T_OFF);
  _Float16* GT     = (_Float16*)(ws + GT_OFF);
  unsigned* Apk    = (unsigned*)(ws + APK_OFF);

  // 0) zero accumulators + scalar maxima (Hacc and Oacc are contiguous)
  k_zero<<<256, 256, 0, stream>>>(Hacc, HACC_N + OACC_N, scal);

  // 1) bit-pack Aq = (A > 0): 256 MB read once; 8 MB packed, L2-resident
  k_pack<<<2048, 256, 0, stream>>>(A, Apk, (long)NN * NN);

  // 2) abs-max scales for X, W_in, W_out
  k_absmax<<<512, 256, 0, stream>>>(X,  NN * DIN,  &scal[0]);
  k_absmax<<<8,   256, 0, stream>>>(Wi, DIN * DH,  &scal[1]);
  k_absmax<<<4,   256, 0, stream>>>(Wo, DH * DOUT, &scal[2]);

  // 3) P1T = (Xq_int @ Wi_int)^T as f16 integers
  k_p1t<<<NN, DH, 0, stream>>>(X, Wi, scal, P1T);

  // 4) Hacc = Aq @ P1int  (exact f32 integers via WMMA f16)
  k_gemm1<<<(NN / 16) * KS1 / 8, 256, 0, stream>>>(Apk, P1T, Hacc);

  // 5) max|Hacc| -> scal[3] (max|h| = max|Hacc| * s1, monotone under RNE mul)
  k_absmax<<<256, 256, 0, stream>>>(Hacc, HACC_N, &scal[3]);

  // 6) GT = (hq_int @ Wo_int)^T as f16 integers
  k_gt<<<NN, DH, 0, stream>>>(Hacc, Wo, scal, GT);

  // 7) Oacc = Aq @ Gint
  k_gemm2<<<(NN / 16) * KS2 / 8, 256, 0, stream>>>(Apk, GT, Oacc);

  // 8) out = Oacc * (s_h * s_wo)
  k_finish<<<256, 256, 0, stream>>>(Oacc, scal, out, OACC_N);
}
